// BertReshapeAttention_78752520339944
// MI455X (gfx1250) — matrix operand order repaired and checked by reading, not compiled, not run on hardware
//
#include <hip/hip_runtime.h>
#include <hip/hip_fp16.h>

// ---------------------------------------------------------------------------
// Slot-attention BERT block for MI455X (gfx1250), wave32 + WMMA f16 (K=32).
// Pipeline:
//   1) convert hs / caches to f16, transpose weights to f16 (B^T layout)
//   2) q,k,v = hs @ W + b           (WMMA GEMM, per-head f16 output)
//   3) pre-attention vs cache_k/v per (b,h)  -> sav_flat  (fused softmax)
//   4) new_key = sav@Wek+bek ; ev = sav@Wev+bev (WMMA GEMM)
//   5) fused final attention: 16 queries x 1008 keys/values per (n,h)
// Matmul inner loops use ping-pong double buffering (manual 2x unroll):
// even/odd K-steps consume statically distinct buffer sets, so prefetch
// loads never WAR-conflict with in-flight WMMA sources and no register
// rotation moves are generated.
// ---------------------------------------------------------------------------

typedef _Float16 half_t;
typedef __attribute__((ext_vector_type(2)))  _Float16 v2h;
typedef __attribute__((ext_vector_type(16))) _Float16 v16h;
typedef __attribute__((ext_vector_type(8)))  float    v8f;

#define NUM_HEADS 12
#define HEAD_DIM  64
#define D_MODEL   768
#define S_SLOT    30
#define B_BATCH   8
#define L_SEQ     16
#define SEQ_CACHE 512
#define N_TOT     (S_SLOT * B_BATCH)   // 240
#define T_SLOT    (S_SLOT * L_SEQ)     // 480
#define KREAL     1008                 // 480 + 512 + 16
#define KPAD      1024

__device__ __forceinline__ v8f wmma_f16(v16h a, v16h b, v8f c) {
  // v_wmma_f32_16x16x32_f16 : C(16x16,f32) += A(16x32,f16) * B(32x16,f16)
  return __builtin_amdgcn_wmma_f32_16x16x32_f16(false, a, false, b, (short)0, c,
                                                false, false);
}

// A-matrix 16x32 f16: lanes 0-15 -> M=0..15 (lanes 16-31 same M),
// VGPR j holds K pair; K = 16*(lane>>4) + 2*j + {0,1} (16 contiguous K per
// lane, matching load_b_tile).
__device__ __forceinline__ v16h load_a_tile(const half_t* a, int lda, int lane) {
  int m  = lane & 15;
  int hh = lane >> 4;
  const half_t* row = a + (size_t)m * lda;
  v16h r;
#pragma unroll
  for (int j = 0; j < 8; ++j) {
    int base = 2 * j;
    int k    = base + 16 * hh;
    v2h pr = *(const v2h*)(row + k);
    r[2 * j]     = pr[0];
    r[2 * j + 1] = pr[1];
  }
  return r;
}

// B-matrix 32x16 f16: lane -> N = lane&15, K-half = lane>>4; 16 contiguous K
// values per lane.  `bt` is B^T row-major: element (k,n) = bt[n*ldb + k].
__device__ __forceinline__ v16h load_b_tile(const half_t* bt, int ldb, int lane) {
  int n   = lane & 15;
  int khf = lane >> 4;
  return *(const v16h*)(bt + (size_t)n * ldb + 16 * khf);
}

// ---------------------------------------------------------------------------
__global__ void cvt_f16_kernel(const float* __restrict__ src,
                               half_t* __restrict__ dst, int n) {
  int i = blockIdx.x * 256 + threadIdx.x;
  if (i < n) dst[i] = (half_t)src[i];
}

__global__ void cvt_transpose768_kernel(const float* __restrict__ w,
                                        half_t* __restrict__ wt) {
  int i = blockIdx.x * 256 + threadIdx.x;      // i over 768*768
  int n = i / D_MODEL, k = i - n * D_MODEL;    // wt[n][k] = w[k][n]
  wt[i] = (half_t)w[(size_t)k * D_MODEL + n];
}

// ---------------------------------------------------------------------------
// C[16x64] tile of  out = A(Mx768,f16) @ W(768x768) + bias, written into the
// per-head layout out[batch][head][row_in_batch][64] as f16.
// K loop: 64 per iteration, ping-pong buffers (even/odd 32-wide K-steps).
__global__ void __launch_bounds__(32)
gemm_head_proj(const half_t* __restrict__ A, const half_t* __restrict__ WT,
               const float* __restrict__ bias, half_t* __restrict__ out,
               int rows_per_batch) {
  int lane = threadIdx.x;
  int r0   = blockIdx.x * 16;
  int head = blockIdx.y;
  int c0   = head * HEAD_DIM;
  const half_t* Abase = A + (size_t)r0 * D_MODEL;

  v8f acc[4] = {};
  v16h ae, be[4];         // even buffers
  v16h ao, bo[4];         // odd buffers
  ae = load_a_tile(Abase, D_MODEL, lane);
#pragma unroll
  for (int t = 0; t < 4; ++t)
    be[t] = load_b_tile(WT + (size_t)(c0 + t * 16) * D_MODEL, D_MODEL, lane);

  for (int k0 = 0; k0 < D_MODEL; k0 += 64) {
    // prefetch odd step (k0+32 <= 736 always in range)
    ao = load_a_tile(Abase + k0 + 32, D_MODEL, lane);
#pragma unroll
    for (int t = 0; t < 4; ++t)
      bo[t] = load_b_tile(WT + (size_t)(c0 + t * 16) * D_MODEL + k0 + 32, D_MODEL, lane);
#pragma unroll
    for (int t = 0; t < 4; ++t)
      acc[t] = wmma_f16(ae, be[t], acc[t]);
    if (k0 + 64 < D_MODEL) {           // prefetch next even step
      ae = load_a_tile(Abase + k0 + 64, D_MODEL, lane);
#pragma unroll
      for (int t = 0; t < 4; ++t)
        be[t] = load_b_tile(WT + (size_t)(c0 + t * 16) * D_MODEL + k0 + 64, D_MODEL, lane);
    }
#pragma unroll
    for (int t = 0; t < 4; ++t)
      acc[t] = wmma_f16(ao, bo[t], acc[t]);
  }

  int nn = lane & 15;
  int mbase = (lane >> 4) * 8;
#pragma unroll
  for (int t = 0; t < 4; ++t) {
#pragma unroll
    for (int r = 0; r < 8; ++r) {
      int g = r0 + mbase + r;
      int batch = g / rows_per_batch;
      int rib   = g - batch * rows_per_batch;
      int col   = c0 + t * 16 + nn;
      out[(size_t)((batch * NUM_HEADS + head) * rows_per_batch + rib) * HEAD_DIM +
          (t * 16 + nn)] = (half_t)(acc[t][r] + bias[col]);
    }
  }
}

// ---------------------------------------------------------------------------
// Stage B: per (s,h,b): softmax(q(16x64)@cache_k^T(64x512)*scale + mask) @ cache_v
// -> sav_flat[b*480 + s*16 + l][h*64 + d]  (f16).  LDS: S 32KB | P 16KB | VT 4KB.
__global__ void __launch_bounds__(32)
preattn_kernel(const half_t* __restrict__ qh, const half_t* __restrict__ ckh,
               const half_t* __restrict__ cvh, const float* __restrict__ amask,
               half_t* __restrict__ sav, float scale) {
  extern __shared__ char smem[];
  float*  S  = (float*)smem;                               // 16 x 512 f32
  half_t* P  = (half_t*)(smem + 16 * 512 * 4);             // 16 x 512 f16
  half_t* VT = (half_t*)(smem + 16 * 512 * 4 + 16 * 512 * 2); // 64 x 32 f16
  int s = blockIdx.x, h = blockIdx.y, b = blockIdx.z;
  int lane = threadIdx.x;
  int n_idx = s * B_BATCH + b;
  const half_t* Q = qh  + (size_t)(n_idx * NUM_HEADS + h) * L_SEQ * HEAD_DIM;
  const half_t* K = ckh + (size_t)(b * NUM_HEADS + h) * SEQ_CACHE * HEAD_DIM;
  const half_t* V = cvh + (size_t)(b * NUM_HEADS + h) * SEQ_CACHE * HEAD_DIM;
  const float* mask = amask + (size_t)b * (SEQ_CACHE + L_SEQ); // ini_mask row b
  int nn = lane & 15, mbase = (lane >> 4) * 8;

  v16h a0 = load_a_tile(Q, HEAD_DIM, lane);        // K 0..31
  v16h a1 = load_a_tile(Q + 32, HEAD_DIM, lane);   // K 32..63

  // scores: ping-pong over key tiles (32 tiles, even/odd pairs)
  v16h e0 = load_b_tile(K, HEAD_DIM, lane);
  v16h e1 = load_b_tile(K + 32, HEAD_DIM, lane);
  v16h o0, o1;
  for (int jt = 0; jt < SEQ_CACHE / 16; jt += 2) {
    const half_t* Ko = K + (size_t)(jt + 1) * 16 * HEAD_DIM;   // always in range
    o0 = load_b_tile(Ko, HEAD_DIM, lane);
    o1 = load_b_tile(Ko + 32, HEAD_DIM, lane);
    v8f c = {};
    c = wmma_f16(a0, e0, c);
    c = wmma_f16(a1, e1, c);
    int col = jt * 16 + nn;
#pragma unroll
    for (int r = 0; r < 8; ++r)
      S[(mbase + r) * SEQ_CACHE + col] = c[r] * scale + mask[col];
    if (jt + 2 < SEQ_CACHE / 16) {
      const half_t* Ke = K + (size_t)(jt + 2) * 16 * HEAD_DIM;
      e0 = load_b_tile(Ke, HEAD_DIM, lane);
      e1 = load_b_tile(Ke + 32, HEAD_DIM, lane);
    }
    v8f c2 = {};
    c2 = wmma_f16(a0, o0, c2);
    c2 = wmma_f16(a1, o1, c2);
    int col2 = col + 16;
#pragma unroll
    for (int r = 0; r < 8; ++r)
      S[(mbase + r) * SEQ_CACHE + col2] = c2[r] * scale + mask[col2];
  }
  __syncthreads();

  for (int m = 0; m < L_SEQ; ++m) {
    float mx = -3.0e38f;
    for (int k2 = lane; k2 < SEQ_CACHE; k2 += 32) mx = fmaxf(mx, S[m * SEQ_CACHE + k2]);
#pragma unroll
    for (int off = 16; off > 0; off >>= 1) mx = fmaxf(mx, __shfl_xor(mx, off, 32));
    float sm = 0.f;
    for (int k2 = lane; k2 < SEQ_CACHE; k2 += 32) {
      float e = __expf(S[m * SEQ_CACHE + k2] - mx);
      S[m * SEQ_CACHE + k2] = e;
      sm += e;
    }
#pragma unroll
    for (int off = 16; off > 0; off >>= 1) sm += __shfl_xor(sm, off, 32);
    float inv = 1.0f / sm;
    for (int k2 = lane; k2 < SEQ_CACHE; k2 += 32)
      P[m * SEQ_CACHE + k2] = (half_t)(S[m * SEQ_CACHE + k2] * inv);
  }
  __syncthreads();

  v8f acc[4] = {};
  for (int kt = 0; kt < SEQ_CACHE / 32; ++kt) {
    const half_t* vr = V + (size_t)(kt * 32 + lane) * HEAD_DIM;
#pragma unroll
    for (int d = 0; d < HEAD_DIM; ++d) VT[d * 32 + lane] = vr[d]; // transpose stage
    __syncthreads();
    v16h pa = load_a_tile(P + kt * 32, SEQ_CACHE, lane);
#pragma unroll
    for (int t = 0; t < 4; ++t)
      acc[t] = wmma_f16(pa, load_b_tile(VT + t * 16 * 32, 32, lane), acc[t]);
    __syncthreads();
  }
#pragma unroll
  for (int t = 0; t < 4; ++t)
#pragma unroll
    for (int r = 0; r < 8; ++r) {
      int row = b * T_SLOT + s * L_SEQ + mbase + r;
      int col = h * HEAD_DIM + t * 16 + nn;
      sav[(size_t)row * D_MODEL + col] = (half_t)acc[t][r];
    }
}

// ---------------------------------------------------------------------------
// Final fused attention per (n,h): keys = [new_key 480 | cache_k 512 | k 16],
// values = [ev 480 | cache_v 512 | v 16], padded to 1024.  Exactly 64KB LDS:
// S = 16x1024 f32; P (16x1024 f16) aliases S rows 8..15 (written descending,
// register-buffered per row); VT (64x32 f16) aliases S row 0 (dead then).
__global__ void __launch_bounds__(32)
final_attn_kernel(const half_t* __restrict__ qh, const half_t* __restrict__ kbuf,
                  const half_t* __restrict__ vbuf, const half_t* __restrict__ nk,
                  const half_t* __restrict__ ev, const half_t* __restrict__ ckh,
                  const half_t* __restrict__ cvh, const float* __restrict__ smask,
                  const float* __restrict__ amask, float* __restrict__ out,
                  float scale) {
  extern __shared__ char smem[];
  float*  S  = (float*)smem;              // 16 x 1024 f32 (64KB)
  half_t* P  = (half_t*)(smem + 32768);   // 16 x 1024 f16 (aliases S rows 8..15)
  half_t* VT = (half_t*)smem;             // 64 x 32 f16 (aliases S row 0)
  int n_idx = blockIdx.x, h = blockIdx.y, lane = threadIdx.x;
  int b = n_idx % B_BATCH;                // n = s*8 + b
  int nn = lane & 15, khf = lane >> 4, mbase = khf * 8;
  const half_t* Q = qh + (size_t)(n_idx * NUM_HEADS + h) * L_SEQ * HEAD_DIM;
  v16h a0 = load_a_tile(Q, HEAD_DIM, lane);
  v16h a1 = load_a_tile(Q + 32, HEAD_DIM, lane);

  // per-lane segmented key-row pointer (j = key index for this lane's column)
  auto key_row = [&](int j) -> const half_t* {
    if (j < T_SLOT)
      return nk + (size_t)((b * NUM_HEADS + h) * T_SLOT + j) * HEAD_DIM;
    if (j < T_SLOT + SEQ_CACHE)
      return ckh + (size_t)((b * NUM_HEADS + h) * SEQ_CACHE + (j - T_SLOT)) * HEAD_DIM;
    if (j < KREAL)
      return kbuf + (size_t)((n_idx * NUM_HEADS + h) * L_SEQ + (j - 992)) * HEAD_DIM;
    return nullptr;
  };
  auto mask_at = [&](int j) -> float {
    if (j < T_SLOT) return smask[(size_t)n_idx * T_SLOT + j];
    if (j < KREAL)  return amask[(size_t)n_idx * (SEQ_CACHE + L_SEQ) + (j - T_SLOT)];
    return 0.f;
  };

  // ---- scores: 16 x 1024, ping-pong over key tiles (64 tiles) ----
  v16h e0 = {}, e1 = {}, o0, o1;
  {
    const half_t* kr = key_row(nn);
    if (kr) {
      e0 = *(const v16h*)(kr + 16 * khf);
      e1 = *(const v16h*)(kr + 32 + 16 * khf);
    }
  }
  for (int jt = 0; jt < KPAD / 16; jt += 2) {
    int j = jt * 16 + nn;
    o0 = (v16h){}; o1 = (v16h){};
    {
      const half_t* kro = key_row(j + 16);
      if (kro) {
        o0 = *(const v16h*)(kro + 16 * khf);
        o1 = *(const v16h*)(kro + 32 + 16 * khf);
      }
    }
    v8f c = {};
    c = wmma_f16(a0, e0, c);
    c = wmma_f16(a1, e1, c);
    float madd = mask_at(j);
#pragma unroll
    for (int r = 0; r < 8; ++r)
      S[(mbase + r) * KPAD + j] = (j < KREAL) ? (c[r] * scale + madd) : -3.0e38f;
    if (jt + 2 < KPAD / 16) {
      e0 = (v16h){}; e1 = (v16h){};
      const half_t* kre = key_row(j + 32);
      if (kre) {
        e0 = *(const v16h*)(kre + 16 * khf);
        e1 = *(const v16h*)(kre + 32 + 16 * khf);
      }
    }
    int j2 = j + 16;
    v8f c2 = {};
    c2 = wmma_f16(a0, o0, c2);
    c2 = wmma_f16(a1, o1, c2);
    float madd2 = mask_at(j2);
#pragma unroll
    for (int r = 0; r < 8; ++r)
      S[(mbase + r) * KPAD + j2] = (j2 < KREAL) ? (c2[r] * scale + madd2) : -3.0e38f;
  }
  __syncthreads();

  // ---- softmax (pass 1: max + exp + sum) ----
  float rinv[16];
  for (int m = 0; m < L_SEQ; ++m) {
    float mx = -3.0e38f;
    for (int k2 = lane; k2 < KPAD; k2 += 32) mx = fmaxf(mx, S[m * KPAD + k2]);
#pragma unroll
    for (int off = 16; off > 0; off >>= 1) mx = fmaxf(mx, __shfl_xor(mx, off, 32));
    float sm = 0.f;
    for (int k2 = lane; k2 < KPAD; k2 += 32) {
      float e = __expf(S[m * KPAD + k2] - mx);
      S[m * KPAD + k2] = e;
      sm += e;
    }
#pragma unroll
    for (int off = 16; off > 0; off >>= 1) sm += __shfl_xor(sm, off, 32);
    rinv[m] = 1.0f / sm;
  }
  __syncthreads();
  // ---- pass 2: S(f32) -> P(f16), descending rows; P aliases S rows >= row m
  for (int m = L_SEQ - 1; m >= 0; --m) {
    float ebuf[32];
#pragma unroll
    for (int i = 0; i < 32; ++i) ebuf[i] = S[m * KPAD + lane + 32 * i];
    __syncthreads();
#pragma unroll
    for (int i = 0; i < 32; ++i)
      P[m * KPAD + lane + 32 * i] = (half_t)(ebuf[i] * rinv[m]);
    __syncthreads();
  }

  // ---- context: P(16x1024) @ V(1024x64) ----
  v8f acc[4] = {};
  for (int kt = 0; kt < KPAD / 32; ++kt) {
    int j = kt * 32 + lane;
    const half_t* vr = nullptr;
    if (j < T_SLOT)
      vr = ev + (size_t)((b * NUM_HEADS + h) * T_SLOT + j) * HEAD_DIM;
    else if (j < T_SLOT + SEQ_CACHE)
      vr = cvh + (size_t)((b * NUM_HEADS + h) * SEQ_CACHE + (j - T_SLOT)) * HEAD_DIM;
    else if (j < KREAL)
      vr = vbuf + (size_t)((n_idx * NUM_HEADS + h) * L_SEQ + (j - 992)) * HEAD_DIM;
    if (vr) {
#pragma unroll
      for (int d = 0; d < HEAD_DIM; ++d) VT[d * 32 + lane] = vr[d];
    } else {
#pragma unroll
      for (int d = 0; d < HEAD_DIM; ++d) VT[d * 32 + lane] = (half_t)0.f;
    }
    __syncthreads();
    v16h pa = load_a_tile(P + kt * 32, KPAD, lane);
#pragma unroll
    for (int t = 0; t < 4; ++t)
      acc[t] = wmma_f16(pa, load_b_tile(VT + t * 16 * 32, 32, lane), acc[t]);
    __syncthreads();
  }
#pragma unroll
  for (int t = 0; t < 4; ++t)
#pragma unroll
    for (int r = 0; r < 8; ++r)
      out[(size_t)(n_idx * L_SEQ + mbase + r) * D_MODEL + h * HEAD_DIM + t * 16 + nn] =
          acc[t][r];
}

// ---------------------------------------------------------------------------
extern "C" void kernel_launch(void* const* d_in, const int* in_sizes, int n_in,
                              void* d_out, int out_size, void* d_ws, size_t ws_size,
                              hipStream_t stream) {
  (void)in_sizes; (void)n_in; (void)out_size; (void)ws_size;
  const float* hs     = (const float*)d_in[0];
  const float* amask  = (const float*)d_in[1];
  const float* smaskp = (const float*)d_in[2];
  const float* cachek = (const float*)d_in[3];
  const float* cachev = (const float*)d_in[4];
  const float* Wq  = (const float*)d_in[5];   const float* bq  = (const float*)d_in[6];
  const float* Wk  = (const float*)d_in[7];   const float* bk  = (const float*)d_in[8];
  const float* Wv  = (const float*)d_in[9];   const float* bv  = (const float*)d_in[10];
  const float* Wek = (const float*)d_in[11];  const float* bek = (const float*)d_in[12];
  const float* Wev = (const float*)d_in[13];  const float* bev = (const float*)d_in[14];
  float* out = (float*)d_out;

  const int HS = N_TOT * L_SEQ * D_MODEL;           // 2,949,120
  const int WE = D_MODEL * D_MODEL;                 // 589,824
  const int CK = B_BATCH * NUM_HEADS * SEQ_CACHE * HEAD_DIM; // 3,145,728

  half_t* p = (half_t*)d_ws;
  half_t* hs_h = p; p += HS;
  half_t* wtq  = p; p += WE;
  half_t* wtk  = p; p += WE;
  half_t* wtv  = p; p += WE;
  half_t* wtek = p; p += WE;
  half_t* wtev = p; p += WE;
  half_t* qh   = p; p += HS;
  half_t* kh   = p; p += HS;
  half_t* vh   = p; p += HS;
  half_t* ckh  = p; p += CK;
  half_t* cvh  = p; p += CK;
  half_t* sav  = p; p += HS;   // [3840][768]
  half_t* nkh  = p; p += HS;   // [8][12][480][64]
  half_t* evh  = p; p += HS;   // [8][12][480][64]

  const float scale = 0.125f;  // 1/sqrt(64)

  cvt_f16_kernel<<<(HS + 255) / 256, 256, 0, stream>>>(hs, hs_h, HS);
  cvt_f16_kernel<<<(CK + 255) / 256, 256, 0, stream>>>(cachek, ckh, CK);
  cvt_f16_kernel<<<(CK + 255) / 256, 256, 0, stream>>>(cachev, cvh, CK);
  cvt_transpose768_kernel<<<WE / 256, 256, 0, stream>>>(Wq,  wtq);
  cvt_transpose768_kernel<<<WE / 256, 256, 0, stream>>>(Wk,  wtk);
  cvt_transpose768_kernel<<<WE / 256, 256, 0, stream>>>(Wv,  wtv);
  cvt_transpose768_kernel<<<WE / 256, 256, 0, stream>>>(Wek, wtek);
  cvt_transpose768_kernel<<<WE / 256, 256, 0, stream>>>(Wev, wtev);

  dim3 gg(N_TOT, NUM_HEADS);   // 3840/16 row tiles x 12 heads
  gemm_head_proj<<<gg, 32, 0, stream>>>(hs_h, wtq, bq, qh, L_SEQ);
  gemm_head_proj<<<gg, 32, 0, stream>>>(hs_h, wtk, bk, kh, L_SEQ);
  gemm_head_proj<<<gg, 32, 0, stream>>>(hs_h, wtv, bv, vh, L_SEQ);

  dim3 gp(S_SLOT, NUM_HEADS, B_BATCH);
  size_t sm_pre = 16 * 512 * 4 + 16 * 512 * 2 + 64 * 32 * 2;  // 53,248 B
  preattn_kernel<<<gp, 32, sm_pre, stream>>>(qh, ckh, cvh, amask, sav, scale);

  gemm_head_proj<<<gg, 32, 0, stream>>>(sav, wtek, bek, nkh, T_SLOT);
  gemm_head_proj<<<gg, 32, 0, stream>>>(sav, wtev, bev, evh, T_SLOT);

  dim3 gf(N_TOT, NUM_HEADS);
  size_t sm_fin = 16 * KPAD * 4;                               // 65,536 B
  final_attn_kernel<<<gf, 32, sm_fin, stream>>>(qh, kh, vh, nkh, evh, ckh, cvh,
                                                smaskp, amask, out, scale);
}
